// ReKT_8589934592386
// MI455X (gfx1250) — matrix operand order repaired and checked by reading, not compiled, not run on hardware
//
#include <hip/hip_runtime.h>
#include <hip/hip_bf16.h>

#define S_LEN   199
#define D_DIM   128
#define N_PRO   20000
#define N_SKILL 500
#define BATCH   512
#define M_ROWS  16
#define NBLK    (BATCH / M_ROWS)

typedef __attribute__((ext_vector_type(16))) __bf16 v16bf;
typedef __attribute__((ext_vector_type(8)))  float  v8f;

__device__ __forceinline__ unsigned short f2bf_bits(float x) {
    union { float f; unsigned u; } v; v.f = x;
    unsigned u = v.u;
    unsigned r = u + 0x7FFFu + ((u >> 16) & 1u);   // round-to-nearest-even
    return (unsigned short)(r >> 16);
}
__device__ __forceinline__ float bf2f_bits(unsigned short h) {
    union { float f; unsigned u; } v; v.u = ((unsigned)h) << 16;
    return v.f;
}
__device__ __forceinline__ float sigmoidf_(float x) {
    return 1.0f / (1.0f + __expf(-x));
}

union AFragU { v16bf v; unsigned u[8]; };
union AccU   { v8f  v; float   f[8]; };

// A fragment (16x32 bf16, MxK) from an LDS row of K-contiguous bf16.
// lanes 0-15 (hi=0): K = 0..7, 16..23 ; lanes 16-31 (hi=1): K = 8..15, 24..31
__device__ __forceinline__ v16bf load_frag_a_lds(const unsigned* row32, int kt, int hi) {
    AFragU f;
    int o = kt * 16 + (hi << 2);
    f.u[0] = row32[o + 0]; f.u[1] = row32[o + 1];
    f.u[2] = row32[o + 2]; f.u[3] = row32[o + 3];
    f.u[4] = row32[o + 8]; f.u[5] = row32[o + 9];
    f.u[6] = row32[o + 10]; f.u[7] = row32[o + 11];
    return f.v;
}
// B fragment: pre-packed per-lane 16 contiguous bf16 (32B aligned) in global ws.
__device__ __forceinline__ v16bf load_frag_b(const unsigned short* p) {
    return *(const v16bf*)p;
}
__device__ __forceinline__ v8f wmma_bf16(v16bf a, v16bf b, v8f c) {
    return __builtin_amdgcn_wmma_f32_16x16x32_bf16(false, a, false, b, (short)0, c, false, false);
}

// ---------------- Prep kernels ----------------

// Zero last-seen tables; init slot-0 of state histories (only slot read pre-write).
__global__ void rekt_prep_init(const float* __restrict__ pro_state0,
                               const float* __restrict__ skill_state0,
                               unsigned short* __restrict__ pro_ws,
                               unsigned short* __restrict__ skill_ws,
                               unsigned char* __restrict__ lpt_tab,
                               unsigned char* __restrict__ lst_tab) {
    long long stride = (long long)gridDim.x * blockDim.x;
    long long i0 = (long long)blockIdx.x * blockDim.x + threadIdx.x;
    for (long long i = i0; i < (long long)BATCH * N_PRO; i += stride)   lpt_tab[i] = 0;
    for (long long i = i0; i < (long long)BATCH * N_SKILL; i += stride) lst_tab[i] = 0;
    for (long long i = i0; i < (long long)BATCH * D_DIM; i += stride) {
        int b = (int)(i >> 7), j = (int)(i & 127);
        pro_ws[(long long)(b * S_LEN) * D_DIM + j]   = f2bf_bits(pro_state0[j]);
        skill_ws[(long long)(b * S_LEN) * D_DIM + j] = f2bf_bits(skill_state0[j]);
    }
}

// Pack W[K][128] f32 -> bf16 WMMA B-fragment order: [kt][nt][lane][e],
// k = kt*32 + (lane>>4)*8 + (e&7) + (e>>3)*16, n = nt*16 + (lane&15).
__global__ void rekt_prep_pack(const float* __restrict__ W,
                               unsigned short* __restrict__ out, int ktiles) {
    int idx = blockIdx.x * blockDim.x + threadIdx.x;
    int total = ktiles * 8 * 512;
    if (idx >= total) return;
    int e    = idx & 15;
    int lane = (idx >> 4) & 31;
    int nt   = (idx >> 9) & 7;
    int kt   = idx >> 12;
    int k = kt * 32 + ((lane >> 4) << 3) + (e & 7) + ((e >> 3) << 4);
    int n = nt * 16 + (lane & 15);
    out[idx] = f2bf_bits(W[k * D_DIM + n]);
}

// ---------------- Main persistent kernel ----------------

__global__ __launch_bounds__(256, 1)
void rekt_main(const int* __restrict__ nxt_p, const int* __restrict__ nxt_s,
               const int* __restrict__ nxt_a,
               const float* __restrict__ pro_embed, const float* __restrict__ skill_embed,
               const float* __restrict__ ans_embed,  const float* __restrict__ time_embed,
               const float* __restrict__ ls_state,
               const float* __restrict__ akt_pro_diff, const float* __restrict__ akt_pro_change,
               const float* __restrict__ b_pf, const float* __restrict__ b_sf,
               const float* __restrict__ b_af, const float* __restrict__ b_ps,
               const float* __restrict__ b_ss, const float* __restrict__ b_as,
               const float* __restrict__ b_o1, const float* __restrict__ W_o2,
               const float* __restrict__ b_o2,
               const unsigned short* __restrict__ wpk,
               unsigned short* __restrict__ pro_ws, unsigned short* __restrict__ skill_ws,
               unsigned char* __restrict__ lpt_tab, unsigned char* __restrict__ lst_tab,
               float* __restrict__ out) {
    // bf16 A-staging: 3 matrices of [16][256] (stride 264) or one [16][512] (stride 520)
    __shared__ unsigned short sStage[3 * 16 * 264];
    __shared__ float sLbp[16 * 132];
    __shared__ float sLbs[16 * 132];
    __shared__ float sAll[16 * 132];   // persistent all_state across steps
    __shared__ float sNpe[16 * 132];   // npe, later reused for relu(hidden)
    __shared__ float sNX [16 * 132];
    __shared__ int sPid[16], sSid[16], sAid[16], sLpt[16], sLst[16];

    const int tid  = threadIdx.x;
    const int lane = tid & 31;
    const int wave = tid >> 5;
    const int hi   = lane >> 4;
    const int lrow = lane & 15;
    const int rr   = tid >> 4;        // row 0..15 for 16-wide staging work
    const int rowBase = blockIdx.x * M_ROWS;

    const unsigned short* wpf = wpk;
    const unsigned short* wsf = wpk + 32768;
    const unsigned short* waf = wpk + 65536;
    const unsigned short* wps = wpk + 98304;
    const unsigned short* wss = wpk + 131072;
    const unsigned short* was = wpk + 163840;
    const unsigned short* wo1 = wpk + 196608;

    // init all_state = ls_state broadcast
    {
        int c0 = (tid & 15) * 8;
        for (int j = c0; j < c0 + 8; ++j) sAll[rr * 132 + j] = ls_state[j];
    }
    __syncthreads();

    for (int t = 0; t < S_LEN; ++t) {
        // ---- Phase A0: per-row scalar gathers + last-seen table update ----
        if (tid < 16) {
            int r = rowBase + tid;
            int p = nxt_p[r * S_LEN + t];
            int s = nxt_s[r * S_LEN + t];
            int a = nxt_a[r * S_LEN + t];
            sPid[tid] = p; sSid[tid] = s; sAid[tid] = a;
            sLpt[tid] = (int)lpt_tab[r * N_PRO + p];
            sLst[tid] = (int)lst_tab[r * N_SKILL + s];
            lpt_tab[r * N_PRO + p]   = (unsigned char)t;
            lst_tab[r * N_SKILL + s] = (unsigned char)t;
        }
        __syncthreads();
        // ---- Phase A1: vector gathers + stage gate-GEMM A operands (bf16) ----
        {
            int r  = rowBase + rr;
            int p  = sPid[rr], s = sSid[rr], a = sAid[rr];
            int lp = sLpt[rr], lsn = sLst[rr];
            float diff = akt_pro_diff[p];
            int c0 = (tid & 15) * 8;
            const unsigned short* lbpRow = pro_ws   + (long long)(r * S_LEN + lp)  * D_DIM;
            const unsigned short* lbsRow = skill_ws + (long long)(r * S_LEN + lsn) * D_DIM;
            const float* tgeRow  = time_embed + (t - lp)  * D_DIM;
            const float* stgeRow = time_embed + (t - lsn) * D_DIM;
            unsigned short* st0 = sStage + (0 * 16 + rr) * 264;
            unsigned short* st1 = sStage + (1 * 16 + rr) * 264;
            unsigned short* st2 = sStage + (2 * 16 + rr) * 264;
            for (int j = c0; j < c0 + 8; ++j) {
                float npe = pro_embed[p * D_DIM + j] + skill_embed[s * D_DIM + j]
                          + diff * akt_pro_change[s * D_DIM + j];
                sNpe[rr * 132 + j] = npe;
                sNX [rr * 132 + j] = npe + ans_embed[a * D_DIM + j];
                unsigned short hb = lbpRow[j];
                sLbp[rr * 132 + j] = bf2f_bits(hb);
                st0[j]       = hb;
                st0[128 + j] = f2bf_bits(tgeRow[j]);
                unsigned short hs = lbsRow[j];
                sLbs[rr * 132 + j] = bf2f_bits(hs);
                st1[j]       = hs;
                st1[128 + j] = f2bf_bits(stgeRow[j]);
                st2[j]       = f2bf_bits(sAll[rr * 132 + j]);
                st2[128 + j] = f2bf_bits(time_embed[D_DIM + j]);  // time_embed[1]
            }
        }
        __syncthreads();
        // ---- Phase B: three forget-gate GEMMs [16,256]x[256,128], sigmoid*state ----
        {
            for (int g = 0; g < 3; ++g) {
                const unsigned short* wp = (g == 0) ? wpf : (g == 1) ? wsf : waf;
                const float* bg          = (g == 0) ? b_pf : (g == 1) ? b_sf : b_af;
                int nt = wave;
                const unsigned* arow = (const unsigned*)(sStage + (g * 16 + lrow) * 264);
                v8f acc = {};
                for (int kt = 0; kt < 8; ++kt) {
                    v16bf aF = load_frag_a_lds(arow, kt, hi);
                    v16bf bF = load_frag_b(wp + ((kt * 8 + nt) * 32 + lane) * 16);
                    acc = wmma_bf16(aF, bF, acc);
                }
                int n = nt * 16 + lrow;
                float bn = bg[n];
                AccU A; A.v = acc;
                float* fb = (g == 0) ? sLbp : (g == 1) ? sLbs : sAll;
                #pragma unroll
                for (int v = 0; v < 8; ++v) {
                    int row = v + (hi << 3);
                    fb[row * 132 + n] *= sigmoidf_(A.f[v] + bn);
                }
            }
        }
        __syncthreads();
        // ---- Phase C1: stage head A = [la | lbp | lbs | npe] as [16,512] bf16 ----
        {
            int c0 = (tid & 15) * 32;
            unsigned short* st = sStage + rr * 520;
            for (int j = c0; j < c0 + 32; ++j) {
                float v;
                if (j < 128)      v = sAll[rr * 132 + j];
                else if (j < 256) v = sLbp[rr * 132 + (j - 128)];
                else if (j < 384) v = sLbs[rr * 132 + (j - 256)];
                else              v = sNpe[rr * 132 + (j - 384)];
                st[j] = f2bf_bits(v);
            }
        }
        __syncthreads();
        // ---- Phase C2: head GEMM [16,512]x[512,128] + relu -> sNpe ----
        {
            int nt = wave;
            const unsigned* arow = (const unsigned*)(sStage + lrow * 520);
            v8f acc = {};
            for (int kt = 0; kt < 16; ++kt) {
                v16bf aF = load_frag_a_lds(arow, kt, hi);
                v16bf bF = load_frag_b(wo1 + ((kt * 8 + nt) * 32 + lane) * 16);
                acc = wmma_bf16(aF, bF, acc);
            }
            int n = nt * 16 + lrow;
            float bn = b_o1[n];
            AccU A; A.v = acc;
            #pragma unroll
            for (int v = 0; v < 8; ++v) {
                int row = v + (hi << 3);
                sNpe[row * 132 + n] = fmaxf(A.f[v] + bn, 0.0f);
            }
        }
        __syncthreads();
        // ---- Phase C3: P = sigmoid(h . W_o2 + b_o2)  (and D1 staging in parallel) ----
        if (tid < 16) {
            float acc = b_o2[0];
            for (int j = 0; j < 128; ++j) acc += sNpe[tid * 132 + j] * W_o2[j];
            out[(rowBase + tid) * S_LEN + t] = sigmoidf_(acc);
        }
        // ---- Phase D1: stage update GEMM A operands [la|nX], [lbp|nX], [lbs|nX] ----
        {
            int c0 = (tid & 15) * 16;
            for (int g = 0; g < 3; ++g) {
                const float* src = (g == 0) ? sAll : (g == 1) ? sLbp : sLbs;
                unsigned short* st = sStage + (g * 16 + rr) * 264;
                for (int j = c0; j < c0 + 16; ++j) {
                    float v = (j < 128) ? src[rr * 132 + j] : sNX[rr * 132 + (j - 128)];
                    st[j] = f2bf_bits(v);
                }
            }
        }
        __syncthreads();
        // ---- Phase D2: update GEMMs + tanh residual; scatter new states ----
        {
            for (int g = 0; g < 3; ++g) {
                const unsigned short* wp = (g == 0) ? was : (g == 1) ? wps : wss;
                const float* bg          = (g == 0) ? b_as : (g == 1) ? b_ps : b_ss;
                int nt = wave;
                const unsigned* arow = (const unsigned*)(sStage + (g * 16 + lrow) * 264);
                v8f acc = {};
                for (int kt = 0; kt < 8; ++kt) {
                    v16bf aF = load_frag_a_lds(arow, kt, hi);
                    v16bf bF = load_frag_b(wp + ((kt * 8 + nt) * 32 + lane) * 16);
                    acc = wmma_bf16(aF, bF, acc);
                }
                int n = nt * 16 + lrow;
                float bn = bg[n];
                AccU A; A.v = acc;
                #pragma unroll
                for (int v = 0; v < 8; ++v) {
                    int row = v + (hi << 3);
                    float th = tanhf(A.f[v] + bn);
                    if (g == 0) {
                        sAll[row * 132 + n] += th;   // new_all = la + tanh(...)
                    } else {
                        const float* base = (g == 1) ? sLbp : sLbs;
                        unsigned short hv = f2bf_bits(base[row * 132 + n] + th);
                        unsigned short* dst = (g == 1) ? pro_ws : skill_ws;
                        dst[(long long)((rowBase + row) * S_LEN + t) * D_DIM + n] = hv;
                    }
                }
            }
        }
        __threadfence();   // make state scatters visible to next step's gathers
        __syncthreads();
    }
}

// ---------------- Launcher ----------------

extern "C" void kernel_launch(void* const* d_in, const int* in_sizes, int n_in,
                              void* d_out, int out_size, void* d_ws, size_t ws_size,
                              hipStream_t stream) {
    const int*   nxt_p          = (const int*)d_in[3];
    const int*   nxt_s          = (const int*)d_in[4];
    const int*   nxt_a          = (const int*)d_in[5];
    const float* pro_embed      = (const float*)d_in[6];
    const float* skill_embed    = (const float*)d_in[7];
    const float* ans_embed      = (const float*)d_in[8];
    const float* time_embed     = (const float*)d_in[9];
    const float* ls_state       = (const float*)d_in[10];
    const float* pro_state0     = (const float*)d_in[11];
    const float* skill_state0   = (const float*)d_in[12];
    const float* akt_pro_diff   = (const float*)d_in[13];
    const float* akt_pro_change = (const float*)d_in[14];
    const float* W_pf = (const float*)d_in[15]; const float* b_pf = (const float*)d_in[16];
    const float* W_sf = (const float*)d_in[17]; const float* b_sf = (const float*)d_in[18];
    const float* W_af = (const float*)d_in[19]; const float* b_af = (const float*)d_in[20];
    const float* W_ps = (const float*)d_in[21]; const float* b_ps = (const float*)d_in[22];
    const float* W_ss = (const float*)d_in[23]; const float* b_ss = (const float*)d_in[24];
    const float* W_as = (const float*)d_in[25]; const float* b_as = (const float*)d_in[26];
    const float* W_o1 = (const float*)d_in[27]; const float* b_o1 = (const float*)d_in[28];
    const float* W_o2 = (const float*)d_in[29]; const float* b_o2 = (const float*)d_in[30];

    // Workspace layout (all 256B-aligned chunks)
    char* ws = (char*)d_ws;
    size_t off = 0;
    unsigned short* pro_ws = (unsigned short*)(ws + off);
    off += (size_t)BATCH * S_LEN * D_DIM * 2;                       // 26,083,328
    unsigned short* skill_ws = (unsigned short*)(ws + off);
    off += (size_t)BATCH * S_LEN * D_DIM * 2;
    unsigned char* lpt_tab = (unsigned char*)(ws + off);
    off += (size_t)BATCH * N_PRO;                                   // 10,240,000
    unsigned char* lst_tab = (unsigned char*)(ws + off);
    off += (size_t)BATCH * N_SKILL;                                 // 256,000
    unsigned short* wpk = (unsigned short*)(ws + off);
    off += (size_t)(6 * 32768 + 65536) * 2;                         // packed weights

    // Prep: zero tables + init slot-0 states
    rekt_prep_init<<<512, 256, 0, stream>>>(pro_state0, skill_state0,
                                            pro_ws, skill_ws, lpt_tab, lst_tab);
    // Prep: pack weights to bf16 B-fragment layout
    rekt_prep_pack<<<128, 256, 0, stream>>>(W_pf, wpk + 0,      8);
    rekt_prep_pack<<<128, 256, 0, stream>>>(W_sf, wpk + 32768,  8);
    rekt_prep_pack<<<128, 256, 0, stream>>>(W_af, wpk + 65536,  8);
    rekt_prep_pack<<<128, 256, 0, stream>>>(W_ps, wpk + 98304,  8);
    rekt_prep_pack<<<128, 256, 0, stream>>>(W_ss, wpk + 131072, 8);
    rekt_prep_pack<<<128, 256, 0, stream>>>(W_as, wpk + 163840, 8);
    rekt_prep_pack<<<256, 256, 0, stream>>>(W_o1, wpk + 196608, 16);

    // Main persistent recurrence: 32 blocks x 256 threads, t-loop inside
    rekt_main<<<NBLK, 256, 0, stream>>>(nxt_p, nxt_s, nxt_a,
                                        pro_embed, skill_embed, ans_embed, time_embed,
                                        ls_state, akt_pro_diff, akt_pro_change,
                                        b_pf, b_sf, b_af, b_ps, b_ss, b_as,
                                        b_o1, W_o2, b_o2,
                                        wpk, pro_ws, skill_ws, lpt_tab, lst_tab,
                                        (float*)d_out);
}
